// W8Linear_15753940041868
// MI455X (gfx1250) — matrix operand order repaired and checked by reading, not compiled, not run on hardware
//
#include <hip/hip_runtime.h>

typedef __attribute__((ext_vector_type(16))) _Float16 v16h;
typedef __attribute__((ext_vector_type(8)))  _Float16 v8h;
typedef __attribute__((ext_vector_type(8)))  float    v8f;
typedef __attribute__((ext_vector_type(4)))  int      v4i;

#define IN_F   4096
#define OUT_F  16384
#define TOKENS 32
#define KSPLIT 4
#define KSPAN  (IN_F / KSPLIT)                       // 1024 K per wave
#define TILES_PER_BLOCK 2
#define WAVES_PER_BLOCK (TILES_PER_BLOCK * KSPLIT)   // 8 waves
#define THREADS (WAVES_PER_BLOCK * 32)               // 256 threads

union V16H { v16h v; v8h h[2]; };

__global__ __launch_bounds__(THREADS)
void w8a16_wmma_kernel(const _Float16* __restrict__ x,
                       const int*      __restrict__ w,
                       const _Float16* __restrict__ scale,
                       const _Float16* __restrict__ bias,
                       float*          __restrict__ out) {
    // deterministic cross-wave K reduction buffer: [wave][c-slot][lane]
    __shared__ float red[WAVES_PER_BLOCK][16][32];

    const int tid   = threadIdx.x;
    const int lane  = tid & 31;
    const int wid   = tid >> 5;
    const int split = wid & (KSPLIT - 1);      // K-split index 0..3
    const int tib   = wid / KSPLIT;            // tile-in-block 0..1
    const int n0    = (blockIdx.x * TILES_PER_BLOCK + tib) * 16;
    const int col   = lane & 15;               // N (and M) sub-index
    const int hi    = lane >> 4;               // half-wave select

    const int kbase = split * KSPAN;

    // A k order chosen here (contiguous, matches B): lanes0-15: K=0..15; lanes16-31: K=16..31
    const _Float16* a0p = x + (col     ) * IN_F + hi * 16 + kbase;   // tokens 0-15
    const _Float16* a1p = x + (col + 16) * IN_F + hi * 16 + kbase;   // tokens 16-31
    // B k order chosen here (contiguous, matches A): lanes0-15: K=0..15; lanes16-31: K=16..31
    const int*      bp  = w + (n0 + col) * IN_F + hi * 16 + kbase;

    v8f c0 = {};   // tokens 0-15  x outputs n0..n0+15
    v8f c1 = {};   // tokens 16-31 x outputs n0..n0+15

    #pragma unroll 2
    for (int k = 0; k < KSPAN; k += 32) {
        V16H a0, a1;
        a0.h[0] = *(const v8h*)(a0p + k);
        a0.h[1] = *(const v8h*)(a0p + k + 8);
        a1.h[0] = *(const v8h*)(a1p + k);
        a1.h[1] = *(const v8h*)(a1p + k + 8);

        // 16 contiguous int32 weights per lane, streamed non-temporally
        v4i wq[4];
        #pragma unroll
        for (int j = 0; j < 4; ++j)
            wq[j] = __builtin_nontemporal_load((const v4i*)(bp + k) + j);

        // exact int8-range -> fp16 convert (|w| <= 127 < 2048)
        v16h b;
        #pragma unroll
        for (int i = 0; i < 16; ++i)
            b[i] = (_Float16)wq[i >> 2][i & 3];

        c0 = __builtin_amdgcn_wmma_f32_16x16x32_f16(false, a0.v, false, b,
                                                    (short)0, c0, false, false);
        c1 = __builtin_amdgcn_wmma_f32_16x16x32_f16(false, a1.v, false, b,
                                                    (short)0, c1, false, false);
    }

    // stash partials in LDS
    #pragma unroll
    for (int r = 0; r < 8; ++r) {
        red[wid][r][lane]     = c0[r];
        red[wid][r + 8][lane] = c1[r];
    }
    __syncthreads();

    // split==0 wave of each tile reduces the 4 K-partials (fixed order ->
    // bitwise deterministic) and applies per-channel scale + bias.
    if (split == 0) {
        const int   n  = n0 + col;
        const float s  = (float)scale[n];
        const float bb = (float)bias[n];
        #pragma unroll
        for (int r = 0; r < 16; ++r) {
            float v = red[wid][r][lane] + red[wid + 1][r][lane]
                    + red[wid + 2][r][lane] + red[wid + 3][r][lane];
            // C layout: VGPR r -> M = r + 8*hi (c1 adds 16)
            const int m = (r & 7) + hi * 8 + (r >> 3) * 16;
            __builtin_nontemporal_store(v * s + bb, out + m * OUT_F + n);
        }
    }
}

extern "C" void kernel_launch(void* const* d_in, const int* in_sizes, int n_in,
                              void* d_out, int out_size, void* d_ws, size_t ws_size,
                              hipStream_t stream) {
    (void)in_sizes; (void)n_in; (void)out_size; (void)d_ws; (void)ws_size;
    const _Float16* x     = (const _Float16*)d_in[0];
    const int*      w     = (const int*)d_in[1];
    const _Float16* scale = (const _Float16*)d_in[2];
    const _Float16* bias  = (const _Float16*)d_in[3];
    float*          out   = (float*)d_out;

    const int n_tiles = OUT_F / 16;                       // 1024
    dim3 grid(n_tiles / TILES_PER_BLOCK);                 // 512 blocks, 4096 waves
    w8a16_wmma_kernel<<<grid, THREADS, 0, stream>>>(x, w, scale, bias, out);
}
